// DynamicFusionModule_18485539242869
// MI455X (gfx1250) — matrix operand order repaired and checked by reading, not compiled, not run on hardware
//
#include <hip/hip_runtime.h>
#include <stdint.h>

#define DIMC   256
#define NTOK   2304
#define NB     4
#define NHEAD  4
#define DH     64
#define FFN    1024
#define QKVD   768
#define MROWS  (NB * NTOK)   // 9216
#define AGHID  512

typedef __attribute__((ext_vector_type(16))) __bf16 v16bf;
typedef __attribute__((ext_vector_type(8)))  float  v8f;

union Frag16 {
    v16bf          v;
    unsigned short u[16];
    uint32_t       w[8];
};
union Acc8 {
    v8f   v;
    float f[8];
};

__device__ __forceinline__ unsigned short f2bf(float f) {
    uint32_t u = __float_as_uint(f);
    uint32_t r = (u + 0x7FFFu + ((u >> 16) & 1u)) >> 16;
    return (unsigned short)r;
}
__device__ __forceinline__ float bf2f(unsigned short h) {
    return __uint_as_float((uint32_t)h << 16);
}

__device__ __forceinline__ v8f wmma_bf16(const Frag16& a, const Frag16& b, v8f c) {
    return __builtin_amdgcn_wmma_f32_16x16x32_bf16(false, a.v, false, b.v,
                                                   (short)0, c, false, false);
}

// ---------------------------------------------------------------------------
// Tensor Data Mover support (gfx1250). Fallback to manual staging if absent.
// ---------------------------------------------------------------------------
#if defined(__gfx1250__) && __has_builtin(__builtin_amdgcn_tensor_load_to_lds) && \
    __has_builtin(__builtin_amdgcn_s_wait_tensorcnt)
#define USE_TDM 1
#else
#define USE_TDM 0
#endif

#if USE_TDM
typedef unsigned int u32x4 __attribute__((ext_vector_type(4)));
typedef int          i32x4 __attribute__((ext_vector_type(4)));
typedef int          i32x8 __attribute__((ext_vector_type(8)));

struct TdmDesc { u32x4 g0; i32x8 g1; i32x4 g2; i32x4 g3; };

__device__ __forceinline__ unsigned lds_addr_of(const void* p) {
    return (unsigned)(unsigned long long)(const __attribute__((address_space(3))) void*)p;
}

// 2D tile descriptor: bf16 elements (data_size=2B), tile [tile1 rows x tile0 cols],
// row stride = stride0 elements, LDS padding (pad_amount+1) DWORDs every
// 2<<pad_interval DWORDs.
__device__ __forceinline__ TdmDesc
make_tdm_2d(unsigned lds_addr, const void* gptr,
            unsigned td0, unsigned td1, unsigned tile0, unsigned tile1,
            unsigned long long stride0, int pad_interval, int pad_amount)
{
    const unsigned long long ga = (unsigned long long)(uintptr_t)gptr;
    TdmDesc d;
    d.g0.x = 1u;                                            // count=1, user mode
    d.g0.y = lds_addr;                                      // lds_addr [63:32]
    d.g0.z = (unsigned)(ga & 0xFFFFFFFFull);                // global_addr lo
    d.g0.w = (unsigned)((ga >> 32) & 0x01FFFFFFu) | (2u << 30); // addr hi | type=2
    unsigned d0 = (1u << 16)                                // data_size = 2 bytes
                | (1u << 20)                                // pad_enable
                | ((unsigned)pad_interval << 22)
                | ((unsigned)pad_amount << 25);
    d.g1[0] = (int)d0;
    d.g1[1] = (int)((td0 & 0xFFFFu) << 16);                 // tensor_dim0 [79:48]
    d.g1[2] = (int)((td0 >> 16) | ((td1 & 0xFFFFu) << 16)); // tensor_dim1 [111:80]
    d.g1[3] = (int)((td1 >> 16) | (tile0 << 16));           // tile_dim0 [127:112]
    d.g1[4] = (int)tile1;                                   // tile_dim1, tile_dim2=0
    d.g1[5] = (int)(unsigned)(stride0 & 0xFFFFFFFFull);     // dim0 stride [207:160]
    d.g1[6] = (int)(unsigned)((stride0 >> 32) & 0xFFFFull);
    d.g1[7] = 0;                                            // dim1 stride unused (2D)
    d.g2[0] = d.g2[1] = d.g2[2] = d.g2[3] = 0;
    d.g3[0] = d.g3[1] = d.g3[2] = d.g3[3] = 0;
    return d;
}

__device__ __forceinline__ void tdm_issue(const TdmDesc& d) {
#if __clang_major__ >= 23
    i32x8 z8 = {0, 0, 0, 0, 0, 0, 0, 0};
    __builtin_amdgcn_tensor_load_to_lds(d.g0, d.g1, d.g2, d.g3, z8, 0);
#else
    __builtin_amdgcn_tensor_load_to_lds(d.g0, d.g1, d.g2, d.g3, 0);
#endif
}
#endif // USE_TDM

// ---------------------------------------------------------------------------
// 0) f32 -> bf16 preconversion (weights; once, outside GEMM hot loops)
// ---------------------------------------------------------------------------
__global__ __launch_bounds__(256)
void cvt_bf16_kernel(const float* __restrict__ src, unsigned short* __restrict__ dst, int n)
{
    const int i = blockIdx.x * 256 + threadIdx.x;
    if (i < n) dst[i] = f2bf(src[i]);
}

// ---------------------------------------------------------------------------
// 1a) Pack [B,C,N] feature maps into row-major [B*N, 512] bf16 (ir | vis).
//     64x64 LDS tile transpose; reads and writes both coalesced.
// ---------------------------------------------------------------------------
__global__ __launch_bounds__(256)
void pack_xcat_kernel(const float* __restrict__ src, unsigned short* __restrict__ xcat,
                      int cofs)
{
    __shared__ unsigned short t[64][65];
    const int b  = blockIdx.x;
    const int n0 = blockIdx.y * 64;
    const int c0 = blockIdx.z * 64;
    const int tx = threadIdx.x & 63, ty = threadIdx.x >> 6;
    for (int i = ty; i < 64; i += 4)
        t[i][tx] = f2bf(src[(size_t)b * DIMC * NTOK + (size_t)(c0 + i) * NTOK + n0 + tx]);
    __syncthreads();
    for (int i = ty; i < 64; i += 4)
        xcat[(size_t)(b * NTOK + n0 + i) * (2 * DIMC) + cofs + c0 + tx] = t[tx][i];
}

// ---------------------------------------------------------------------------
// 1b) Agent logits: per-pixel dot(h1[pix], aw2) + ab2, one wave per pixel.
// ---------------------------------------------------------------------------
__global__ __launch_bounds__(256)
void agent_logits_kernel(const unsigned short* __restrict__ h1,
                         const float* __restrict__ aw2, const float* __restrict__ ab2,
                         float* __restrict__ logits)
{
    const int wave = threadIdx.x >> 5, lane = threadIdx.x & 31;
    const int pix = blockIdx.x * 8 + wave;
    const unsigned short* hp = h1 + (size_t)pix * AGHID;
    float acc = 0.f;
    for (int i = lane; i < AGHID; i += 32)
        acc += bf2f(hp[i]) * aw2[i];
    for (int d = 1; d < 32; d <<= 1) acc += __shfl_xor(acc, d, 32);
    if (lane == 0) logits[pix] = acc + ab2[0];
}

// ---------------------------------------------------------------------------
// 2) Selection mask + top-64 fallback (one block per batch)
// ---------------------------------------------------------------------------
__global__ __launch_bounds__(256)
void select_kernel(const float* __restrict__ logits, int* __restrict__ sel,
                   float* __restrict__ maskf)
{
    const int b = blockIdx.x, tid = threadIdx.x;
    __shared__ float lcp[NTOK];
    __shared__ float rv[256];
    __shared__ int   ri[256];

    int cnt = 0;
    for (int i = tid; i < NTOK; i += 256) {
        float v = logits[b * NTOK + i];
        lcp[i] = v;
        int m = (v > 0.f) ? 1 : 0;
        sel[b * NTOK + i] = m;
        maskf[b * NTOK + i] = (float)m;
        cnt += m;
    }
    ri[tid] = cnt;
    __syncthreads();
    for (int s = 128; s > 0; s >>= 1) {
        if (tid < s) ri[tid] += ri[tid + s];
        __syncthreads();
    }
    const int total = ri[0];
    __syncthreads();
    if (total >= 64) return;

    for (int i = tid; i < NTOK; i += 256) sel[b * NTOK + i] = 0;
    __syncthreads();
    for (int it = 0; it < 64; ++it) {
        float bm = -1e30f; int bi = NTOK;
        for (int i = tid; i < NTOK; i += 256) {
            float v = lcp[i];
            if (v > bm || (v == bm && i < bi)) { bm = v; bi = i; }
        }
        rv[tid] = bm; ri[tid] = bi;
        __syncthreads();
        for (int s = 128; s > 0; s >>= 1) {
            if (tid < s) {
                if (rv[tid + s] > rv[tid] ||
                    (rv[tid + s] == rv[tid] && ri[tid + s] < ri[tid])) {
                    rv[tid] = rv[tid + s]; ri[tid] = ri[tid + s];
                }
            }
            __syncthreads();
        }
        if (tid == 0) { sel[b * NTOK + ri[0]] = 1; lcp[ri[0]] = -1e38f; }
        __syncthreads();
    }
}

// ---------------------------------------------------------------------------
// 3) LayerNorm -> bf16 rows.  strided==1: x[b,n,c] = f[b*C*N + c*N + n]
// ---------------------------------------------------------------------------
__global__ __launch_bounds__(256)
void ln_kernel(const float* __restrict__ x, int strided,
               const float* __restrict__ g, const float* __restrict__ beta,
               unsigned short* __restrict__ out)
{
    const int row = blockIdx.x, c = threadIdx.x;
    __shared__ float sd[256];
    float v;
    if (strided) {
        int b = row / NTOK, n = row % NTOK;
        v = x[(size_t)b * DIMC * NTOK + (size_t)c * NTOK + n];
    } else {
        v = x[(size_t)row * DIMC + c];
    }
    sd[c] = v;
    __syncthreads();
    for (int s = 128; s > 0; s >>= 1) { if (c < s) sd[c] += sd[c + s]; __syncthreads(); }
    const float mean = sd[0] * (1.f / 256.f);
    __syncthreads();
    const float d = v - mean;
    sd[c] = d * d;
    __syncthreads();
    for (int s = 128; s > 0; s >>= 1) { if (c < s) sd[c] += sd[c + s]; __syncthreads(); }
    const float var = sd[0] * (1.f / 256.f);
    const float xn = d * rsqrtf(var + 1e-5f) * g[c] + beta[c];
    out[(size_t)row * DIMC + c] = f2bf(xn);
}

// ---------------------------------------------------------------------------
// 4) WMMA bf16 GEMM:  out[M,Nc] = A[M,K](bf16) @ W[Nc,K]^T(bf16) + bias
//    Tile 128x64, 8 waves, 4 WMMA / wave / 32-K step.
//    A/W tiles staged via TDM (double-buffered) when available.
// ---------------------------------------------------------------------------
__global__ __launch_bounds__(256)
void gemm_bf16_kernel(const unsigned short* __restrict__ A,
                      const unsigned short* __restrict__ Wb,
                      const float* __restrict__ bias,
                      int K, int Nc,
                      float* __restrict__ outF, unsigned short* __restrict__ outH,
                      int act,                           // 1 = exact gelu, 2 = silu
                      const float* __restrict__ res, int res_mode, // 0/1 contig/2 f-strided
                      int accum)
{
    __shared__ unsigned short As[2][128 * 40];
    __shared__ unsigned short Ws[2][64 * 40];
    const int tid  = threadIdx.x;
    const int lane = tid & 31, wave = tid >> 5;
    const int lmod = lane & 15, lhalf = lane >> 4;
    const int wm   = wave & 3,  wn    = wave >> 2;
    const int m0   = blockIdx.x * 128, n0 = blockIdx.y * 64;
    const int nk   = K >> 5;

    Acc8 acc[2][2];
    for (int mi = 0; mi < 2; ++mi)
        for (int ni = 0; ni < 2; ++ni)
            for (int r = 0; r < 8; ++r) acc[mi][ni].f[r] = 0.f;

#if USE_TDM
    const unsigned aL0 = lds_addr_of(&As[0][0]), aL1 = lds_addr_of(&As[1][0]);
    const unsigned wL0 = lds_addr_of(&Ws[0][0]), wL1 = lds_addr_of(&Ws[1][0]);
    if (wave == 0) {
        tdm_issue(make_tdm_2d(aL0, A  + (size_t)m0 * K, K, MROWS, 32, 128, K, 3, 3));
        tdm_issue(make_tdm_2d(wL0, Wb + (size_t)n0 * K, K, Nc,    32,  64, K, 3, 3));
    }
#endif

    for (int kc = 0; kc < nk; ++kc) {
#if USE_TDM
        const int buf = kc & 1;
        __syncthreads();   // everyone done reading buf^1 before DMA rewrites it
        if (wave == 0) {
            if (kc + 1 < nk) {
                const int k1 = (kc + 1) << 5;
                tdm_issue(make_tdm_2d(buf ? aL0 : aL1, A  + (size_t)m0 * K + k1,
                                      K, MROWS, 32, 128, K, 3, 3));
                tdm_issue(make_tdm_2d(buf ? wL0 : wL1, Wb + (size_t)n0 * K + k1,
                                      K, Nc, 32, 64, K, 3, 3));
                __builtin_amdgcn_s_wait_tensorcnt(2);  // current tile landed
            } else {
                __builtin_amdgcn_s_wait_tensorcnt(0);
            }
        }
        __syncthreads();
#else
        const int buf = 0;
        const int k0 = kc << 5;
        __syncthreads();
        {
            const int ar = tid >> 1, ac = (tid & 1) << 4;
            const unsigned short* src = A + (size_t)(m0 + ar) * K + k0 + ac;
            *reinterpret_cast<uint4*>(&As[0][ar * 40 + ac])     = *reinterpret_cast<const uint4*>(src);
            *reinterpret_cast<uint4*>(&As[0][ar * 40 + ac + 8]) = *reinterpret_cast<const uint4*>(src + 8);
            const int wr = tid >> 2, wc = (tid & 3) << 3;
            *reinterpret_cast<uint4*>(&Ws[0][wr * 40 + wc]) =
                *reinterpret_cast<const uint4*>(Wb + (size_t)(n0 + wr) * K + k0 + wc);
        }
        __syncthreads();
#endif
        Frag16 af[2];
        for (int mi = 0; mi < 2; ++mi)
            for (int j = 0; j < 8; ++j) {
                const int k = ((j >> 2) << 4) + lhalf * 8 + ((j & 3) << 1);
                af[mi].w[j] = *reinterpret_cast<const uint32_t*>(
                    &As[buf][(wm * 32 + mi * 16 + lmod) * 40 + k]);
            }
        for (int ni = 0; ni < 2; ++ni) {
            Frag16 bf_;
            const int n = wn * 32 + ni * 16 + lmod;
            for (int j = 0; j < 8; ++j)
                bf_.w[j] = *reinterpret_cast<const uint32_t*>(
                    &Ws[buf][n * 40 + ((j >> 2) << 4) + lhalf * 8 + ((j & 3) << 1)]);
            for (int mi = 0; mi < 2; ++mi)
                acc[mi][ni].v = wmma_bf16(af[mi], bf_, acc[mi][ni].v);
        }
    }

    for (int mi = 0; mi < 2; ++mi)
        for (int ni = 0; ni < 2; ++ni) {
            const int n = n0 + wn * 32 + ni * 16 + lmod;
            for (int r = 0; r < 8; ++r) {
                const int m = m0 + wm * 32 + mi * 16 + r + lhalf * 8;
                float v = acc[mi][ni].f[r] + bias[n];
                if (act == 1)      v = 0.5f * v * (1.0f + erff(v * 0.70710678f));
                else if (act == 2) v = v / (1.f + __expf(-v));
                if (res_mode == 1) {
                    v += res[(size_t)m * Nc + n];
                } else if (res_mode == 2) {
                    const int bb = m / NTOK, nt = m % NTOK;
                    v += res[(size_t)bb * DIMC * NTOK + (size_t)n * NTOK + nt];
                }
                if (outH)        outH[(size_t)m * Nc + n] = f2bf(v);
                else if (accum)  outF[(size_t)m * Nc + n] += v;
                else             outF[(size_t)m * Nc + n] = v;
            }
        }
}

// ---------------------------------------------------------------------------
// 5) Flash-style masked attention: one wave per 32-query tile per (b,h).
//    Per 32-key tile: 8 WMMA QK^T + 8 WMMA PV; K/V tiles staged via TDM.
// ---------------------------------------------------------------------------
__global__ __launch_bounds__(32)
void attn_kernel(const unsigned short* __restrict__ qkv, const int* __restrict__ sel,
                 unsigned short* __restrict__ o_attn)
{
    const int qt = blockIdx.x;               // 0..71 (32 queries each)
    const int b  = blockIdx.y >> 2;
    const int h  = blockIdx.y & 3;
    const int lane = threadIdx.x;
    const int lmod = lane & 15, lhalf = lane >> 4;

    __shared__ unsigned short kls[2][32 * 72];
    __shared__ unsigned short vls[2][32 * 72];
    __shared__ unsigned short pls[32 * 32];

    // Q fragments (A layout): 2 query sub-tiles x 2 K-chunks of 32
    Frag16 qf[2][2];
    for (int qi = 0; qi < 2; ++qi) {
        const int tok = qt * 32 + qi * 16 + lmod;
        const unsigned short* qrow = qkv + (size_t)(b * NTOK + tok) * QKVD + h * DH;
        for (int c2 = 0; c2 < 2; ++c2)
            for (int j = 0; j < 8; ++j) {
                const int k = c2 * 32 + ((j >> 2) << 4) + lhalf * 8 + ((j & 3) << 1);
                qf[qi][c2].w[j] = *reinterpret_cast<const uint32_t*>(qrow + k);
            }
    }

    Acc8 O[2][4];
    for (int qi = 0; qi < 2; ++qi)
        for (int f = 0; f < 4; ++f)
            for (int r = 0; r < 8; ++r) O[qi][f].f[r] = 0.f;
    float Mrow[2][8], Lrow[2][8];
    for (int qi = 0; qi < 2; ++qi)
        for (int r = 0; r < 8; ++r) { Mrow[qi][r] = -1e30f; Lrow[qi][r] = 0.f; }

    const int nkt = NTOK / 32;
#if USE_TDM
    const unsigned kA0 = lds_addr_of(&kls[0][0]), kA1 = lds_addr_of(&kls[1][0]);
    const unsigned vA0 = lds_addr_of(&vls[0][0]), vA1 = lds_addr_of(&vls[1][0]);
    {
        const unsigned short* kg = qkv + (size_t)(b * NTOK) * QKVD + DIMC + h * DH;
        tdm_issue(make_tdm_2d(kA0, kg,        QKVD, MROWS, 64, 32, QKVD, 4, 3));
        tdm_issue(make_tdm_2d(vA0, kg + DIMC, QKVD, MROWS, 64, 32, QKVD, 4, 3));
    }
#endif

    for (int kt = 0; kt < nkt; ++kt) {
#if USE_TDM
        const int buf = kt & 1;
        __syncthreads();
        if (kt + 1 < nkt) {
            const unsigned short* kg =
                qkv + (size_t)(b * NTOK + (kt + 1) * 32) * QKVD + DIMC + h * DH;
            tdm_issue(make_tdm_2d(buf ? kA0 : kA1, kg,        QKVD, MROWS, 64, 32, QKVD, 4, 3));
            tdm_issue(make_tdm_2d(buf ? vA0 : vA1, kg + DIMC, QKVD, MROWS, 64, 32, QKVD, 4, 3));
            __builtin_amdgcn_s_wait_tensorcnt(2);
        } else {
            __builtin_amdgcn_s_wait_tensorcnt(0);
        }
        __syncthreads();
#else
        const int buf = 0;
        __syncthreads();
        {
            const int key = kt * 32 + lane;
            const unsigned short* kr =
                qkv + (size_t)(b * NTOK + key) * QKVD + DIMC + h * DH;
            for (int i = 0; i < DH; i += 8) {
                *reinterpret_cast<uint4*>(&kls[0][lane * 72 + i]) =
                    *reinterpret_cast<const uint4*>(kr + i);
                *reinterpret_cast<uint4*>(&vls[0][lane * 72 + i]) =
                    *reinterpret_cast<const uint4*>(kr + DIMC + i);
            }
        }
        __syncthreads();
#endif
        // K fragments (shared by both query sub-tiles)
        Frag16 kf[2][2];
        for (int fi = 0; fi < 2; ++fi)
            for (int c2 = 0; c2 < 2; ++c2)
                for (int j = 0; j < 8; ++j) {
                    const int d = c2 * 32 + ((j >> 2) << 4) + lhalf * 8 + ((j & 3) << 1);
                    kf[fi][c2].w[j] = *reinterpret_cast<const uint32_t*>(
                        &kls[buf][(fi * 16 + lmod) * 72 + d]);
                }

        const bool sel0 = sel[b * NTOK + kt * 32 + lmod] != 0;
        const bool sel1 = sel[b * NTOK + kt * 32 + 16 + lmod] != 0;

        for (int qi = 0; qi < 2; ++qi) {
            Acc8 sc[2];
            for (int f = 0; f < 2; ++f)
                for (int r = 0; r < 8; ++r) sc[f].f[r] = 0.f;
            for (int fi = 0; fi < 2; ++fi)
                for (int c2 = 0; c2 < 2; ++c2)
                    sc[fi].v = wmma_bf16(qf[qi][c2], kf[fi][c2], sc[fi].v);

            for (int r = 0; r < 8; ++r) {
                float s0 = sel0 ? sc[0].f[r] * 0.125f : -1e9f;
                float s1 = sel1 ? sc[1].f[r] * 0.125f : -1e9f;
                float mxr = fmaxf(s0, s1);
                for (int d = 1; d < 16; d <<= 1) mxr = fmaxf(mxr, __shfl_xor(mxr, d, 16));
                const float Mn = fmaxf(Mrow[qi][r], mxr);
                const float alpha = __expf(Mrow[qi][r] - Mn);
                const float p0 = __expf(s0 - Mn);
                const float p1 = __expf(s1 - Mn);
                float ps = p0 + p1;
                for (int d = 1; d < 16; d <<= 1) ps += __shfl_xor(ps, d, 16);
                Lrow[qi][r] = Lrow[qi][r] * alpha + ps;
                Mrow[qi][r] = Mn;
                for (int f = 0; f < 4; ++f) O[qi][f].f[r] *= alpha;
                const int mrow = qi * 16 + r + lhalf * 8;
                pls[mrow * 32 + lmod]      = f2bf(p0);
                pls[mrow * 32 + 16 + lmod] = f2bf(p1);
            }
        }
        __syncthreads();

        Frag16 pa[2];
        for (int qi = 0; qi < 2; ++qi)
            for (int j = 0; j < 8; ++j) {
                const int k = ((j >> 2) << 4) + lhalf * 8 + ((j & 3) << 1);
                pa[qi].w[j] = *reinterpret_cast<const uint32_t*>(
                    &pls[(qi * 16 + lmod) * 32 + k]);
            }

        for (int fn = 0; fn < 4; ++fn) {
            Frag16 vf;
            const int d = fn * 16 + lmod;
            for (int j = 0; j < 8; ++j) {
                const int kk = ((j >> 2) << 4) + lhalf * 8 + ((j & 3) << 1);
                uint32_t lo = vls[buf][kk * 72 + d];
                uint32_t hi = vls[buf][(kk + 1) * 72 + d];
                vf.w[j] = lo | (hi << 16);
            }
            for (int qi = 0; qi < 2; ++qi)
                O[qi][fn].v = wmma_bf16(pa[qi], vf, O[qi][fn].v);
        }
        __syncthreads();
    }

    for (int qi = 0; qi < 2; ++qi)
        for (int fn = 0; fn < 4; ++fn) {
            const int d = fn * 16 + lmod;
            for (int r = 0; r < 8; ++r) {
                const int tok = qt * 32 + qi * 16 + r + lhalf * 8;
                const float v = O[qi][fn].f[r] / Lrow[qi][r];
                o_attn[(size_t)(b * NTOK + tok) * DIMC + h * DH + d] = f2bf(v);
            }
        }
}

// ---------------------------------------------------------------------------
// 6) Scatter-combine into [B,C,H,W] canvas
// ---------------------------------------------------------------------------
__global__ __launch_bounds__(256)
void combine_kernel(const float* __restrict__ f_ir, const float* __restrict__ f_vis,
                    const float* __restrict__ y_sum, const int* __restrict__ sel,
                    const float* __restrict__ maskf, float* __restrict__ out)
{
    const size_t i = (size_t)blockIdx.x * 256 + threadIdx.x;
    if (i >= (size_t)NB * DIMC * NTOK) return;
    const int n = (int)(i % NTOK);
    const size_t t = i / NTOK;
    const int c = (int)(t % DIMC);
    const int b = (int)(t / DIMC);
    const int p = b * NTOK + n;
    float v;
    if (sel[p]) v = y_sum[(size_t)p * DIMC + c] * maskf[p];
    else        v = f_ir[i] + f_vis[i];
    out[i] = v;
}

// ---------------------------------------------------------------------------
extern "C" void kernel_launch(void* const* d_in, const int* in_sizes, int n_in,
                              void* d_out, int out_size, void* d_ws, size_t ws_size,
                              hipStream_t stream)
{
    const float* f_ir  = (const float*)d_in[0];
    const float* f_vis = (const float*)d_in[1];
    const float* aw1   = (const float*)d_in[2];
    const float* ab1   = (const float*)d_in[3];
    const float* aw2   = (const float*)d_in[4];
    const float* ab2   = (const float*)d_in[5];

    struct Mix { const float *lng, *lnb, *wqkv, *bqkv, *wo, *bo, *w1, *b1, *w2, *b2; };
    Mix mix[2];
    for (int m = 0; m < 2; ++m) {
        const int base = 6 + m * 10;
        mix[m].lng  = (const float*)d_in[base + 0];
        mix[m].lnb  = (const float*)d_in[base + 1];
        mix[m].wqkv = (const float*)d_in[base + 2];
        mix[m].bqkv = (const float*)d_in[base + 3];
        mix[m].wo   = (const float*)d_in[base + 4];
        mix[m].bo   = (const float*)d_in[base + 5];
        mix[m].w1   = (const float*)d_in[base + 6];
        mix[m].b1   = (const float*)d_in[base + 7];
        mix[m].w2   = (const float*)d_in[base + 8];
        mix[m].b2   = (const float*)d_in[base + 9];
    }

    char* ws = (char*)d_ws;
    size_t off = 0;
    auto take = [&](size_t bytes) -> void* {
        void* p = ws + off;
        off = (off + bytes + 255) & ~(size_t)255;
        return p;
    };
    float*          logits = (float*)         take((size_t)MROWS * 4);
    int*            sel    = (int*)           take((size_t)MROWS * 4);
    float*          maskf  = (float*)         take((size_t)MROWS * 4);
    unsigned short* xn     = (unsigned short*)take((size_t)MROWS * DIMC * 2);
    unsigned short* qkv    = (unsigned short*)take((size_t)MROWS * QKVD * 2);
    unsigned short* o_att  = (unsigned short*)take((size_t)MROWS * DIMC * 2);
    float*          x1     = (float*)         take((size_t)MROWS * DIMC * 4);
    unsigned short* hbuf   = (unsigned short*)take((size_t)MROWS * FFN  * 2);
    float*          y_sum  = (float*)         take((size_t)MROWS * DIMC * 4);
    unsigned short* xcat   = (unsigned short*)take((size_t)MROWS * AGHID * 2);
    unsigned short* h1     = (unsigned short*)take((size_t)MROWS * AGHID * 2);
    unsigned short* aw1_bf = (unsigned short*)take((size_t)AGHID * AGHID * 2);
    unsigned short* wq_bf[2], *wo_bf[2], *w1_bf[2], *w2_bf[2];
    for (int m = 0; m < 2; ++m) {
        wq_bf[m] = (unsigned short*)take((size_t)QKVD * DIMC * 2);
        wo_bf[m] = (unsigned short*)take((size_t)DIMC * DIMC * 2);
        w1_bf[m] = (unsigned short*)take((size_t)FFN  * DIMC * 2);
        w2_bf[m] = (unsigned short*)take((size_t)DIMC * FFN  * 2);
    }

    auto cvt = [&](const float* src, unsigned short* dst, int n) {
        cvt_bf16_kernel<<<(n + 255) / 256, 256, 0, stream>>>(src, dst, n);
    };
    cvt(aw1, aw1_bf, AGHID * AGHID);
    for (int m = 0; m < 2; ++m) {
        cvt(mix[m].wqkv, wq_bf[m], QKVD * DIMC);
        cvt(mix[m].wo,   wo_bf[m], DIMC * DIMC);
        cvt(mix[m].w1,   w1_bf[m], FFN  * DIMC);
        cvt(mix[m].w2,   w2_bf[m], DIMC * FFN);
    }

    // SamplingAgent: pack -> WMMA GEMM (SiLU) -> rank-1 logits
    pack_xcat_kernel<<<dim3(NB, NTOK / 64, DIMC / 64), 256, 0, stream>>>(f_ir,  xcat, 0);
    pack_xcat_kernel<<<dim3(NB, NTOK / 64, DIMC / 64), 256, 0, stream>>>(f_vis, xcat, DIMC);
    gemm_bf16_kernel<<<dim3(MROWS / 128, AGHID / 64), 256, 0, stream>>>(
        xcat, aw1_bf, ab1, AGHID, AGHID, nullptr, h1, 2, nullptr, 0, 0);
    agent_logits_kernel<<<MROWS / 8, 256, 0, stream>>>(h1, aw2, ab2, logits);
    select_kernel<<<NB, 256, 0, stream>>>(logits, sel, maskf);

    for (int m = 0; m < 2; ++m) {
        const Mix& mp = mix[m];
        const float* f = m ? f_vis : f_ir;

        ln_kernel<<<MROWS, 256, 0, stream>>>(f, 1, mp.lng, mp.lnb, xn);
        gemm_bf16_kernel<<<dim3(MROWS / 128, QKVD / 64), 256, 0, stream>>>(
            xn, wq_bf[m], mp.bqkv, DIMC, QKVD, nullptr, qkv, 0, nullptr, 0, 0);

        attn_kernel<<<dim3(NTOK / 32, NB * NHEAD), 32, 0, stream>>>(qkv, sel, o_att);

        gemm_bf16_kernel<<<dim3(MROWS / 128, DIMC / 64), 256, 0, stream>>>(
            o_att, wo_bf[m], mp.bo, DIMC, DIMC, x1, nullptr, 0, f, 2, 0);

        ln_kernel<<<MROWS, 256, 0, stream>>>(x1, 0, mp.lng, mp.lnb, xn);
        gemm_bf16_kernel<<<dim3(MROWS / 128, FFN / 64), 256, 0, stream>>>(
            xn, w1_bf[m], mp.b1, DIMC, FFN, nullptr, hbuf, 1, nullptr, 0, 0);
        gemm_bf16_kernel<<<dim3(MROWS / 128, DIMC / 64), 256, 0, stream>>>(
            hbuf, w2_bf[m], mp.b2, FFN, DIMC, y_sum, nullptr, 0, x1, 1, m);
    }

    const size_t total = (size_t)NB * DIMC * NTOK;
    combine_kernel<<<(unsigned)((total + 255) / 256), 256, 0, stream>>>(
        f_ir, f_vis, y_sum, sel, maskf, (float*)d_out);
}
